// ExphormerLayer_438086664594
// MI455X (gfx1250) — matrix operand order repaired and checked by reading, not compiled, not run on hardware
//
#include <hip/hip_runtime.h>

#define NNODES 50000
#define NEDGES 1600000
#define DIM 128
#define NH 8
#define HD 16
#define NTILES_N 3125      /* 50000/16  */
#define NTILES_E 100000    /* 1600000/16 */
#define LDB 136            /* padded LDS row stride (bf16) for staged weights */

typedef __attribute__((ext_vector_type(16))) __bf16 v16bf;
typedef __attribute__((ext_vector_type(8)))  float  v8f;
typedef __attribute__((ext_vector_type(4)))  int    v4i;

typedef __attribute__((address_space(1))) v4i* gv4i_p;   // global v4i*
typedef __attribute__((address_space(3))) v4i* lv4i_p;   // LDS v4i*

// ---------------------------------------------------------------------------
// WMMA fragment helpers (layouts per CDNA5 ISA 7.12.2, wave32)
// ---------------------------------------------------------------------------

// A fragment (16x32, bf16) from f32 row-major source, converting on the fly.
// lanes 0-15: row = lane,  elems 0..7 -> K=kb..kb+7,   8..15 -> K=kb+16..kb+23
// lanes16-31: row = lane-16, same but K offset +8 / +24.
__device__ inline v16bf load_A_f32(const float* base, int ld, int kb) {
  const int lane = threadIdx.x & 31;
  const int row  = lane & 15;
  const int kh   = (lane & 16) ? 8 : 0;
  const float* p = base + row * ld + kb + kh;
  v16bf a;
#pragma unroll
  for (int i = 0; i < 8; ++i) a[i] = (__bf16)p[i];
#pragma unroll
  for (int i = 0; i < 8; ++i) a[8 + i] = (__bf16)p[16 + i];
  return a;
}

// B fragment (32x16, bf16) for computing x @ W^T:  B[k][n] = W[n][k].
// lane holds column n = lane&15; elems 0..7 -> K=kb+kh..kb+kh+7, 8..15 ->
// K=kb+16+kh..kb+kh+23, kh = 0 (lanes 0-15) / 8 (lanes 16-31).  W is row-major bf16 with row stride ld.
__device__ inline v16bf load_B_bf16(const __bf16* W, int ld, int ntile, int kb) {
  const int lane = threadIdx.x & 31;
  const int n  = ntile * 16 + (lane & 15);
  const int k0 = kb + ((lane & 16) ? 8 : 0);
  const __bf16* p = W + (size_t)n * ld + k0;
  v16bf b;
#pragma unroll
  for (int i = 0; i < 8; ++i) b[i] = p[i];
#pragma unroll
  for (int i = 0; i < 8; ++i) b[8 + i] = p[16 + i];
  return b;
}

__device__ inline v8f wmma_bf16(v16bf a, v16bf b, v8f c) {
  return __builtin_amdgcn_wmma_f32_16x16x32_bf16(false, a, false, b,
                                                 (short)0, c, false, false);
}

// ---------------------------------------------------------------------------
// Async staging of bf16 weight rows (128 bf16 = 256B each) into padded LDS
// rows of LDB bf16.  Uses the CDNA5 async-to-LDS path when the toolchain
// exposes it; falls back to load+store otherwise.
// ---------------------------------------------------------------------------
#if __has_builtin(__builtin_amdgcn_global_load_async_to_lds_b128)
#define EDGE_ASYNC_STAGE 1
#else
#define EDGE_ASYNC_STAGE 0
#endif

__device__ inline void stage_rows(const __bf16* __restrict__ gW,
                                  __bf16* sW, int rows, int tid, int nthreads) {
  const int chunks = rows * 16;            // 16-byte chunks per 256B row
  for (int ch = tid; ch < chunks; ch += nthreads) {
    int r = ch >> 4, c = ch & 15;
#if EDGE_ASYNC_STAGE
    __builtin_amdgcn_global_load_async_to_lds_b128(
        (gv4i_p)(gW + (size_t)r * DIM + c * 8),
        (lv4i_p)(sW + (size_t)r * LDB + c * 8),
        0, 0);
#else
    const int4 v = *(const int4*)(gW + (size_t)r * DIM + c * 8);
    *(int4*)(sW + (size_t)r * LDB + c * 8) = v;
#endif
  }
}

__device__ inline void stage_wait() {
#if EDGE_ASYNC_STAGE
#if __has_builtin(__builtin_amdgcn_s_wait_asynccnt)
  __builtin_amdgcn_s_wait_asynccnt(0);
#else
  asm volatile("s_wait_asynccnt 0x0" ::: "memory");
#endif
#endif
}

// ---------------------------------------------------------------------------
// Prep / utility kernels
// ---------------------------------------------------------------------------
__global__ void cvt_bf16_kernel(const float* __restrict__ src,
                                __bf16* __restrict__ dst, int n) {
  int i = blockIdx.x * blockDim.x + threadIdx.x;
  if (i < n) dst[i] = (__bf16)src[i];
}

// Pack Wb (8x128) into zero-padded 16x128 bf16 so bias fits one WMMA N-tile.
__global__ void pad_wb_kernel(const float* __restrict__ Wb,
                              __bf16* __restrict__ dst) {
  int i = blockIdx.x * blockDim.x + threadIdx.x;  // 0..2047
  if (i < 16 * DIM) {
    int r = i >> 7;
    dst[i] = (__bf16)((r < NH) ? Wb[i] : 0.0f);
  }
}

__global__ void zero_kernel(float* __restrict__ p, size_t n) {
  size_t i = blockIdx.x * (size_t)blockDim.x + threadIdx.x;
  size_t stride = (size_t)gridDim.x * blockDim.x;
  for (; i < n; i += stride) p[i] = 0.0f;
}

// ---------------------------------------------------------------------------
// Node QKV:  Qh/Kh/Vh = h @ W{q,k,v}^T  (N x 128)
// grid = 3125 node tiles, block = 256 (8 waves), wave w -> N-tile w.
// ---------------------------------------------------------------------------
__global__ __launch_bounds__(256) void qkv_kernel(
    const float* __restrict__ h,
    const __bf16* __restrict__ Wqb, const __bf16* __restrict__ Wkb,
    const __bf16* __restrict__ Wvb,
    float* __restrict__ Qh, float* __restrict__ Kh, float* __restrict__ Vh) {
  const int wave = threadIdx.x >> 5;
  const int lane = threadIdx.x & 31;
  const int tile = blockIdx.x;
  const float* hb = h + (size_t)tile * 16 * DIM;

  v16bf a0 = load_A_f32(hb, DIM, 0);
  v16bf a1 = load_A_f32(hb, DIM, 32);
  v16bf a2 = load_A_f32(hb, DIM, 64);
  v16bf a3 = load_A_f32(hb, DIM, 96);

  v8f q = {}, k = {}, v = {};
  q = wmma_bf16(a0, load_B_bf16(Wqb, DIM, wave, 0),  q);
  q = wmma_bf16(a1, load_B_bf16(Wqb, DIM, wave, 32), q);
  q = wmma_bf16(a2, load_B_bf16(Wqb, DIM, wave, 64), q);
  q = wmma_bf16(a3, load_B_bf16(Wqb, DIM, wave, 96), q);

  k = wmma_bf16(a0, load_B_bf16(Wkb, DIM, wave, 0),  k);
  k = wmma_bf16(a1, load_B_bf16(Wkb, DIM, wave, 32), k);
  k = wmma_bf16(a2, load_B_bf16(Wkb, DIM, wave, 64), k);
  k = wmma_bf16(a3, load_B_bf16(Wkb, DIM, wave, 96), k);

  v = wmma_bf16(a0, load_B_bf16(Wvb, DIM, wave, 0),  v);
  v = wmma_bf16(a1, load_B_bf16(Wvb, DIM, wave, 32), v);
  v = wmma_bf16(a2, load_B_bf16(Wvb, DIM, wave, 64), v);
  v = wmma_bf16(a3, load_B_bf16(Wvb, DIM, wave, 96), v);

  const int col = lane & 15;
  const int rb  = (lane & 16) ? 8 : 0;
#pragma unroll
  for (int j = 0; j < 8; ++j) {
    size_t row = (size_t)tile * 16 + rb + j;
    Qh[row * DIM + wave * 16 + col] = q[j];
    Kh[row * DIM + wave * 16 + col] = k[j];
    Vh[row * DIM + wave * 16 + col] = v[j];
  }
}

// ---------------------------------------------------------------------------
// Edge kernel: per 16-edge tile (one wave each):
//   - Web/Wbp staged block-wide into padded LDS via async-to-LDS
//   - E tile = edge_attr @ We^T (WMMA, B from LDS), bias via padded Wb
//   - score = exp(clip(sum_d K[src]*Q[dst]*E / 4 + bias, -5, 5))
//   - scatter msg = Vh[src]*score into wV, score into Z (f32 atomics)
// grid = 12500 blocks x 8 waves = 100000 tiles.
// ---------------------------------------------------------------------------
__global__ __launch_bounds__(256) void edge_kernel(
    const float* __restrict__ edge_attr, const int* __restrict__ ei,
    const __bf16* __restrict__ Web, const __bf16* __restrict__ Wbp,
    const float* __restrict__ bb,
    const float* __restrict__ Qh, const float* __restrict__ Kh,
    const float* __restrict__ Vh,
    float* __restrict__ wV, float* __restrict__ Z) {
  __shared__ __attribute__((aligned(16))) __bf16 s_web[DIM * LDB];  // 34 KB
  __shared__ __attribute__((aligned(16))) __bf16 s_wbp[16 * LDB];   // 4.25 KB
  __shared__ int   s_src[8][16];
  __shared__ int   s_dst[8][16];
  __shared__ float s_sc[8][16][NH];
  __shared__ float s_bias[8][16][NH];

  const int tid  = threadIdx.x;
  const int wave = tid >> 5;
  const int lane = tid & 31;
  const int tile = blockIdx.x * 8 + wave;
  const int e0   = tile * 16;

  // Block-wide async staging of We (128 rows) + padded Wb (16 rows) into LDS.
  stage_rows(Web, s_web, DIM, tid, 256);
  stage_rows(Wbp, s_wbp, 16, tid, 256);

  // Edge indices + prefetch of the gather rows while the stage is in flight.
  if (lane < 16) {
    int my_src = ei[e0 + lane];
    int my_dst = ei[NEDGES + e0 + lane];
    s_src[wave][lane] = my_src;
    s_dst[wave][lane] = my_dst;
    __builtin_prefetch(Kh + (size_t)my_src * DIM, 0, 1);
    __builtin_prefetch(Qh + (size_t)my_dst * DIM, 0, 1);
    __builtin_prefetch(Vh + (size_t)my_src * DIM, 0, 1);
  }

  stage_wait();
  __syncthreads();

  const float* eb = edge_attr + (size_t)e0 * DIM;
  v16bf a0 = load_A_f32(eb, DIM, 0);
  v16bf a1 = load_A_f32(eb, DIM, 32);
  v16bf a2 = load_A_f32(eb, DIM, 64);
  v16bf a3 = load_A_f32(eb, DIM, 96);

  const int col = lane & 15;
  const int rb  = (lane & 16) ? 8 : 0;

  // E_bias tile (only columns 0..7 meaningful; Wbp rows 8..15 are zero).
  {
    v8f cb = {};
    cb = wmma_bf16(a0, load_B_bf16(s_wbp, LDB, 0, 0),  cb);
    cb = wmma_bf16(a1, load_B_bf16(s_wbp, LDB, 0, 32), cb);
    cb = wmma_bf16(a2, load_B_bf16(s_wbp, LDB, 0, 64), cb);
    cb = wmma_bf16(a3, load_B_bf16(s_wbp, LDB, 0, 96), cb);
    if (col < NH) {
#pragma unroll
      for (int j = 0; j < 8; ++j)
        s_bias[wave][rb + j][col] = cb[j] + bb[col];
    }
  }
  __builtin_amdgcn_wave_barrier();

  // Per head: one 16x16 E tile, then the per-edge attention dot.
  for (int hh = 0; hh < NH; ++hh) {
    v8f c = {};
    c = wmma_bf16(a0, load_B_bf16(s_web, LDB, hh, 0),  c);
    c = wmma_bf16(a1, load_B_bf16(s_web, LDB, hh, 32), c);
    c = wmma_bf16(a2, load_B_bf16(s_web, LDB, hh, 64), c);
    c = wmma_bf16(a3, load_B_bf16(s_web, LDB, hh, 96), c);
#pragma unroll
    for (int j = 0; j < 8; ++j) {
      const int e = rb + j;                 // edge index in tile (per half)
      const int s = s_src[wave][e];
      const int d = s_dst[wave][e];
      float p = Kh[(size_t)s * DIM + hh * HD + col] *
                Qh[(size_t)d * DIM + hh * HD + col] * c[j];
      // reduce over the 16 head dims (lanes within each 16-lane half)
      p += __shfl_xor(p, 8);
      p += __shfl_xor(p, 4);
      p += __shfl_xor(p, 2);
      p += __shfl_xor(p, 1);
      if (col == 0) {  // lanes 0 (e=j) and 16 (e=8+j)
        float sc = p * 0.25f + s_bias[wave][e][hh];   // 1/sqrt(16) = 0.25
        sc = fminf(fmaxf(sc, -5.0f), 5.0f);
        s_sc[wave][e][hh] = __expf(sc);
      }
    }
  }
  __builtin_amdgcn_wave_barrier();

  // Z scatter: 16 edges x 8 heads = 128 adds, 4 passes of 32 lanes.
#pragma unroll
  for (int pass = 0; pass < 4; ++pass) {
    int idx = pass * 32 + lane;
    int e = idx >> 3, hh = idx & 7;
    atomicAdd(&Z[(size_t)s_dst[wave][e] * NH + hh], s_sc[wave][e][hh]);
  }
  // msg scatter: per edge, 128 dims in 4 passes.
  for (int e = 0; e < 16; ++e) {
    const int s = s_src[wave][e];
    const int d = s_dst[wave][e];
#pragma unroll
    for (int pass = 0; pass < 4; ++pass) {
      int dim = pass * 32 + lane;
      float mv = Vh[(size_t)s * DIM + dim] * s_sc[wave][e][dim >> 4];
      atomicAdd(&wV[(size_t)d * DIM + dim], mv);
    }
  }
}

// ---------------------------------------------------------------------------
// x = h + wV / (Z + 1e-6)
// ---------------------------------------------------------------------------
__global__ void combine_kernel(const float* __restrict__ h,
                               const float* __restrict__ wV,
                               const float* __restrict__ Z,
                               float* __restrict__ x) {
  size_t i = blockIdx.x * (size_t)blockDim.x + threadIdx.x;
  if (i >= (size_t)NNODES * DIM) return;
  size_t n = i >> 7;
  int d = (int)(i & 127);
  x[i] = h[i] + wV[i] / (Z[n * NH + (d >> 4)] + 1e-6f);
}

// ---------------------------------------------------------------------------
// BatchNorm column stats: per-column sum & sumsq (block-partial + atomic).
// block = 128 threads (one per column).
// ---------------------------------------------------------------------------
__global__ __launch_bounds__(128) void stats_kernel(
    const float* __restrict__ x, float* __restrict__ sum,
    float* __restrict__ sq, int rows, int rows_per_block) {
  int c  = threadIdx.x;
  int r0 = blockIdx.x * rows_per_block;
  int r1 = min(r0 + rows_per_block, rows);
  float s = 0.0f, q = 0.0f;
  for (int r = r0; r < r1; ++r) {
    float v = x[(size_t)r * DIM + c];
    s += v;
    q += v * v;
  }
  atomicAdd(&sum[c], s);
  atomicAdd(&sq[c], q);
}

__global__ __launch_bounds__(128) void bn_finalize_kernel(
    const float* __restrict__ sum, const float* __restrict__ sq,
    const float* __restrict__ gamma, const float* __restrict__ beta,
    float* __restrict__ scale, float* __restrict__ shift) {
  int c = threadIdx.x;
  const float inv_n = 1.0f / (float)NNODES;
  float mean = sum[c] * inv_n;
  float var  = sq[c] * inv_n - mean * mean;
  float sc   = gamma[c] * rsqrtf(var + 1e-5f);
  scale[c] = sc;
  shift[c] = beta[c] - mean * sc;
}

// ---------------------------------------------------------------------------
// FFN:  xn = BN1(x);  t = xn + relu(xn@W1^T + b1)@W2^T + b2
// block = 64 (2 waves), each wave owns a 16-row tile; per-wave LDS:
//   xn tile (16x128 f32, 8KB) + y tile (16x256 f32, 16KB) = 24KB.
// ---------------------------------------------------------------------------
__global__ __launch_bounds__(64) void ffn_kernel(
    const float* __restrict__ x,
    const float* __restrict__ scale1, const float* __restrict__ shift1,
    const __bf16* __restrict__ W1b, const float* __restrict__ b1,
    const __bf16* __restrict__ W2b, const float* __restrict__ b2,
    float* __restrict__ t) {
  extern __shared__ float smem[];
  const int wave = threadIdx.x >> 5;
  const int lane = threadIdx.x & 31;
  const int tile = blockIdx.x * 2 + wave;
  if (tile >= NTILES_N) return;

  float* xn = smem + wave * (16 * DIM + 16 * 2 * DIM);
  float* y  = xn + 16 * DIM;
  const size_t row0 = (size_t)tile * 16;

  // Load + normalize tile into LDS.
#pragma unroll
  for (int i = 0; i < 64; ++i) {
    int idx = i * 32 + lane;       // 0..2047
    int c = idx & 127;
    xn[idx] = x[row0 * DIM + idx] * scale1[c] + shift1[c];
  }
  __builtin_amdgcn_wave_barrier();

  v16bf a0 = load_A_f32(xn, DIM, 0);
  v16bf a1 = load_A_f32(xn, DIM, 32);
  v16bf a2 = load_A_f32(xn, DIM, 64);
  v16bf a3 = load_A_f32(xn, DIM, 96);

  const int col = lane & 15;
  const int rb  = (lane & 16) ? 8 : 0;

  // GEMM1: y (16x256) = relu(xn @ W1^T + b1)
  for (int nt = 0; nt < 16; ++nt) {
    v8f c0 = {};
    c0 = wmma_bf16(a0, load_B_bf16(W1b, DIM, nt, 0),  c0);
    c0 = wmma_bf16(a1, load_B_bf16(W1b, DIM, nt, 32), c0);
    c0 = wmma_bf16(a2, load_B_bf16(W1b, DIM, nt, 64), c0);
    c0 = wmma_bf16(a3, load_B_bf16(W1b, DIM, nt, 96), c0);
    float bv = b1[nt * 16 + col];
#pragma unroll
    for (int j = 0; j < 8; ++j)
      y[(rb + j) * (2 * DIM) + nt * 16 + col] = fmaxf(c0[j] + bv, 0.0f);
  }
  __builtin_amdgcn_wave_barrier();

  // GEMM2: f (16x128) = y @ W2^T;  t = xn + f + b2
  v16bf ay[8];
#pragma unroll
  for (int kk = 0; kk < 8; ++kk) ay[kk] = load_A_f32(y, 2 * DIM, kk * 32);

  for (int nt = 0; nt < 8; ++nt) {
    v8f c0 = {};
#pragma unroll
    for (int kk = 0; kk < 8; ++kk)
      c0 = wmma_bf16(ay[kk], load_B_bf16(W2b, 2 * DIM, nt, kk * 32), c0);
    int gc = nt * 16 + col;
    float bv = b2[gc];
#pragma unroll
    for (int j = 0; j < 8; ++j) {
      int row = rb + j;
      t[(row0 + row) * DIM + gc] = xn[row * DIM + gc] + c0[j] + bv;
    }
  }
}

// ---------------------------------------------------------------------------
// out = t * scale2 + shift2
// ---------------------------------------------------------------------------
__global__ void apply_bn_kernel(const float* __restrict__ t,
                                const float* __restrict__ scale,
                                const float* __restrict__ shift,
                                float* __restrict__ out) {
  size_t i = blockIdx.x * (size_t)blockDim.x + threadIdx.x;
  if (i >= (size_t)NNODES * DIM) return;
  int c = (int)(i & 127);
  out[i] = t[i] * scale[c] + shift[c];
}

// ---------------------------------------------------------------------------
// Launcher
// ---------------------------------------------------------------------------
static inline size_t align_up(size_t x) { return (x + 255) & ~(size_t)255; }

extern "C" void kernel_launch(void* const* d_in, const int* in_sizes, int n_in,
                              void* d_out, int out_size, void* d_ws,
                              size_t ws_size, hipStream_t stream) {
  (void)in_sizes; (void)n_in; (void)out_size; (void)ws_size;
  const float* h      = (const float*)d_in[0];
  const int*   ei     = (const int*)d_in[1];
  const float* ea     = (const float*)d_in[2];
  const float* Wq     = (const float*)d_in[3];
  const float* Wk     = (const float*)d_in[4];
  const float* Wv     = (const float*)d_in[5];
  const float* We     = (const float*)d_in[6];
  const float* Wb     = (const float*)d_in[7];
  const float* bb     = (const float*)d_in[8];
  const float* gamma1 = (const float*)d_in[9];
  const float* beta1  = (const float*)d_in[10];
  const float* W1     = (const float*)d_in[11];
  const float* b1     = (const float*)d_in[12];
  const float* W2     = (const float*)d_in[13];
  const float* b2     = (const float*)d_in[14];
  const float* gamma2 = (const float*)d_in[15];
  const float* beta2  = (const float*)d_in[16];
  float* out = (float*)d_out;

  char* ws = (char*)d_ws;
  size_t off = 0;
  auto allocf = [&](size_t n) -> float* {
    float* p = (float*)(ws + off);
    off = align_up(off + n * sizeof(float));
    return p;
  };
  auto allocb = [&](size_t n) -> __bf16* {
    __bf16* p = (__bf16*)(ws + off);
    off = align_up(off + n * sizeof(__bf16));
    return p;
  };

  float* Qh  = allocf((size_t)NNODES * DIM);
  float* Kh  = allocf((size_t)NNODES * DIM);
  float* Vh  = allocf((size_t)NNODES * DIM);
  float* wV  = allocf((size_t)NNODES * DIM);
  float* Z   = allocf((size_t)NNODES * NH);
  float* x   = allocf((size_t)NNODES * DIM);
  float* t   = allocf((size_t)NNODES * DIM);
  float* sum1 = allocf(DIM); float* sq1 = allocf(DIM);
  float* sum2 = allocf(DIM); float* sq2 = allocf(DIM);
  float* scale1 = allocf(DIM); float* shift1 = allocf(DIM);
  float* scale2 = allocf(DIM); float* shift2 = allocf(DIM);
  __bf16* Wqb = allocb(DIM * DIM);
  __bf16* Wkb = allocb(DIM * DIM);
  __bf16* Wvb = allocb(DIM * DIM);
  __bf16* Web = allocb(DIM * DIM);
  __bf16* W1b = allocb(2 * DIM * DIM);
  __bf16* W2b = allocb(2 * DIM * DIM);
  __bf16* Wbp = allocb(16 * DIM);

  // --- prep: weight conversion ---
  cvt_bf16_kernel<<<(DIM * DIM + 255) / 256, 256, 0, stream>>>(Wq, Wqb, DIM * DIM);
  cvt_bf16_kernel<<<(DIM * DIM + 255) / 256, 256, 0, stream>>>(Wk, Wkb, DIM * DIM);
  cvt_bf16_kernel<<<(DIM * DIM + 255) / 256, 256, 0, stream>>>(Wv, Wvb, DIM * DIM);
  cvt_bf16_kernel<<<(DIM * DIM + 255) / 256, 256, 0, stream>>>(We, Web, DIM * DIM);
  cvt_bf16_kernel<<<(2 * DIM * DIM + 255) / 256, 256, 0, stream>>>(W1, W1b, 2 * DIM * DIM);
  cvt_bf16_kernel<<<(2 * DIM * DIM + 255) / 256, 256, 0, stream>>>(W2, W2b, 2 * DIM * DIM);
  pad_wb_kernel<<<(16 * DIM + 255) / 256, 256, 0, stream>>>(Wb, Wbp);

  // --- zero accumulators ---
  zero_kernel<<<2048, 256, 0, stream>>>(wV, (size_t)NNODES * DIM);
  zero_kernel<<<256, 256, 0, stream>>>(Z, (size_t)NNODES * NH);
  zero_kernel<<<1, 256, 0, stream>>>(sum1, DIM);
  zero_kernel<<<1, 256, 0, stream>>>(sq1, DIM);
  zero_kernel<<<1, 256, 0, stream>>>(sum2, DIM);
  zero_kernel<<<1, 256, 0, stream>>>(sq2, DIM);

  // --- main pipeline ---
  qkv_kernel<<<NTILES_N, 256, 0, stream>>>(h, Wqb, Wkb, Wvb, Qh, Kh, Vh);

  edge_kernel<<<NTILES_E / 8, 256, 0, stream>>>(ea, ei, Web, Wbp, bb,
                                                Qh, Kh, Vh, wV, Z);

  combine_kernel<<<((size_t)NNODES * DIM + 255) / 256, 256, 0, stream>>>(h, wV, Z, x);

  stats_kernel<<<(NNODES + 127) / 128, 128, 0, stream>>>(x, sum1, sq1, NNODES, 128);
  bn_finalize_kernel<<<1, 128, 0, stream>>>(sum1, sq1, gamma1, beta1, scale1, shift1);

  size_t ffn_lds = 2 * (16 * DIM + 16 * 2 * DIM) * sizeof(float);  // 48 KB
  ffn_kernel<<<(NTILES_N + 1) / 2, 64, ffn_lds, stream>>>(x, scale1, shift1,
                                                          W1b, b1, W2b, b2, t);

  stats_kernel<<<(NNODES + 127) / 128, 128, 0, stream>>>(t, sum2, sq2, NNODES, 128);
  bn_finalize_kernel<<<1, 128, 0, stream>>>(sum2, sq2, gamma2, beta2, scale2, shift2);

  apply_bn_kernel<<<((size_t)NNODES * DIM + 255) / 256, 256, 0, stream>>>(t, scale2, shift2, out);
}
